// GraphEnhancedLSTM_90907277787255
// MI455X (gfx1250) — matrix operand order repaired and checked by reading, not compiled, not run on hardware
//
#include <hip/hip_runtime.h>

// ---- shapes ----
constexpr int Bz = 16, Sz = 512, Fz = 64, Hz = 128;
constexpr int Nz = 1024, GHz = 64, FUSEz = 64, H1z = 32;
constexpr int G4H = 4 * Hz;          // 512 gate columns
constexpr int KT  = Hz + Nz;         // 1152 fused K

// ---- workspace layout (bytes) ----
constexpr size_t OFF_XSEQ = 0;                                     // S*B*F bf16
constexpr size_t OFF_Y0   = OFF_XSEQ + (size_t)Sz * Bz * Fz * 2;   // S*B*H bf16
constexpr size_t OFF_W0IH = OFF_Y0   + (size_t)Sz * Bz * Hz * 2;
constexpr size_t OFF_W0HH = OFF_W0IH + (size_t)G4H * Fz * 2;
constexpr size_t OFF_W1IH = OFF_W0HH + (size_t)G4H * Hz * 2;
constexpr size_t OFF_W1HH = OFF_W1IH + (size_t)G4H * Hz * 2;
constexpr size_t OFF_PR   = OFF_W1HH + (size_t)G4H * Hz * 2;       // B*H f32
constexpr size_t OFF_S    = OFF_PR   + (size_t)Bz * Hz * 4;        // B*N f32
constexpr size_t OFF_WFG  = OFF_S    + (size_t)Bz * Nz * 4;        // N*FUSE f32
constexpr size_t OFF_AB   = OFF_WFG  + (size_t)Nz * FUSEz * 4;     // alpha,beta
constexpr size_t OFF_CA   = OFF_AB   + 256;                        // B*KT bf16
constexpr size_t OFF_BT   = OFF_CA   + (size_t)Bz * KT * 2;        // FUSE*KT bf16

// ---- WMMA types / helpers ----
typedef __attribute__((ext_vector_type(16))) __bf16 v16bf;
typedef __attribute__((ext_vector_type(8)))  float  v8f;
struct PK32 { uint4 a, b; };

__device__ inline v16bf pack32(uint4 lo, uint4 hi) {
  PK32 p{lo, hi};
  return __builtin_bit_cast(v16bf, p);
}

__device__ inline __bf16 f2bf(float x) {
  unsigned u = __builtin_bit_cast(unsigned, x);
  unsigned r = u + 0x7FFFu + ((u >> 16) & 1u);   // round-to-nearest-even
  unsigned short h = (unsigned short)(r >> 16);
  return __builtin_bit_cast(__bf16, h);
}

__device__ inline float sigm(float x) { return 1.f / (1.f + __expf(-x)); }

// A tile 16x32 bf16 (row-major src, M=row, K). Caller passes base+k0.
// lanes 0-15: M=lane, K 0..7 & 16..23 ; lanes 16-31: M=lane-16, K 8..15 & 24..31.
__device__ inline v16bf load_atile(const __bf16* base, int ld, int lane) {
  int row = lane & 15;
  int kb  = (lane >> 4) * 8;
  const __bf16* p = base + row * ld + kb;
  uint4 lo = *(const uint4*)(p);
  uint4 hi = *(const uint4*)(p + 16);
  return pack32(lo, hi);
}

// B tile 32x16 bf16 where B[k,n] = W[n,k], W row-major (n rows, ldk cols).
// lanes 0-15: N=n0+lane, K 0..7 & 16..23 ; lanes 16-31: N=n0+lane-16, K 8..15 & 24..31.
__device__ inline v16bf load_btile(const __bf16* W, int ldk, int n0, int k0, int lane) {
  int col = n0 + (lane & 15);
  int kb  = k0 + ((lane >> 4) << 3);
  const __bf16* p = W + (size_t)col * ldk + kb;
  uint4 lo = *(const uint4*)(p);
  uint4 hi = *(const uint4*)(p + 16);
  return pack32(lo, hi);
}

__device__ inline v8f wmma_bf16(v16bf a, v16bf b, v8f c) {
  return __builtin_amdgcn_wmma_f32_16x16x32_bf16(false, a, false, b, (short)0, c,
                                                 false, false);
}

// ================= prep: bf16 conversions, xseq transpose, alpha/beta ========
__global__ void prep_kernel(const float* __restrict__ price,
                            const float* __restrict__ wih0, const float* __restrict__ whh0,
                            const float* __restrict__ wih1, const float* __restrict__ whh1,
                            const float* __restrict__ Wg, const float* __restrict__ a_src,
                            const float* __restrict__ a_dst,
                            __bf16* xseq, __bf16* w0ih, __bf16* w0hh,
                            __bf16* w1ih, __bf16* w1hh, float* ab) {
  int tid = blockIdx.x * blockDim.x + threadIdx.x;
  int stride = gridDim.x * blockDim.x;
  for (int idx = tid; idx < Sz * Bz * Fz; idx += stride) {   // [t][b][f] <- [b][t][f]
    int t = idx / (Bz * Fz), r = idx % (Bz * Fz);
    int b = r / Fz, f = r % Fz;
    xseq[idx] = f2bf(price[(size_t)b * Sz * Fz + (size_t)t * Fz + f]);
  }
  for (int idx = tid; idx < G4H * Fz; idx += stride) w0ih[idx] = f2bf(wih0[idx]);
  for (int idx = tid; idx < G4H * Hz; idx += stride) w0hh[idx] = f2bf(whh0[idx]);
  for (int idx = tid; idx < G4H * Hz; idx += stride) w1ih[idx] = f2bf(wih1[idx]);
  for (int idx = tid; idx < G4H * Hz; idx += stride) w1hh[idx] = f2bf(whh1[idx]);
  if (blockIdx.x == 0 && threadIdx.x == 0) {
    float a = 0.f, bt = 0.f;
    for (int k = 0; k < GHz; k++) { a += Wg[k] * a_src[k]; bt += Wg[k] * a_dst[k]; }
    ab[0] = a; ab[1] = bt;
  }
}

// ========== WfG[n,f] = sum_k Wg[k] * Wf[128 + n*64 + k, f]  (one pass over Wf) ==
__global__ void wfg_kernel(const float* __restrict__ Wg, const float* __restrict__ Wf,
                           float* __restrict__ WfG) {
  int n = blockIdx.x, f = threadIdx.x;
  float acc = 0.f;
  for (int k = 0; k < GHz; k++)
    acc += Wg[k] * Wf[(size_t)(Hz + n * GHz + k) * FUSEz + f];
  WfG[n * FUSEz + f] = acc;
}

// ================= LSTM: one workgroup, 8 waves, weights in VGPRs ============
template <int KXT>
__device__ void run_layer(const __bf16* __restrict__ xseq,
                          const __bf16* __restrict__ Wih, const __bf16* __restrict__ Whh,
                          const float* __restrict__ bih, const float* __restrict__ bhh,
                          __bf16* __restrict__ yout, float* __restrict__ hfinal,
                          float* gates, float* cst, __bf16* hst, float* bias, int tid) {
  const int KX = KXT * 32;
  int wave = tid >> 5, lane = tid & 31;
  for (int i = tid; i < Bz * Hz; i += 256) {
    cst[i] = 0.f;
    hst[i] = __builtin_bit_cast(__bf16, (unsigned short)0);
  }
  for (int j = tid; j < G4H; j += 256) bias[j] = bih[j] + bhh[j];

  v16bf wx[4][KXT], wh[4][4];                      // register-resident weights
#pragma unroll
  for (int j = 0; j < 4; j++) {
    int n0 = wave * 64 + j * 16;
#pragma unroll
    for (int kt = 0; kt < KXT; kt++) wx[j][kt] = load_btile(Wih, KX, n0, kt * 32, lane);
#pragma unroll
    for (int kt = 0; kt < 4; kt++)   wh[j][kt] = load_btile(Whh, Hz, n0, kt * 32, lane);
  }
  __syncthreads();

  for (int t = 0; t < Sz; t++) {
    const __bf16* xt = xseq + (size_t)t * Bz * KX;
    v16bf ax[KXT], ah[4];
#pragma unroll
    for (int kt = 0; kt < KXT; kt++) ax[kt] = load_atile(xt + kt * 32, KX, lane);
#pragma unroll
    for (int kt = 0; kt < 4; kt++)   ah[kt] = load_atile(hst + kt * 32, Hz, lane);
#pragma unroll
    for (int j = 0; j < 4; j++) {
      v8f acc = {};
#pragma unroll
      for (int kt = 0; kt < KXT; kt++) acc = wmma_bf16(ax[kt], wx[j][kt], acc);
#pragma unroll
      for (int kt = 0; kt < 4; kt++)   acc = wmma_bf16(ah[kt], wh[j][kt], acc);
      int n  = wave * 64 + j * 16 + (lane & 15);
      int mb = (lane >> 4) * 8;
#pragma unroll
      for (int r = 0; r < 8; r++) gates[(mb + r) * G4H + n] = acc[r];
    }
    __syncthreads();
    for (int idx = tid; idx < Bz * Hz; idx += 256) {
      int b = idx >> 7, jj = idx & 127;
      const float* gr = gates + b * G4H;
      float ig = sigm(gr[jj] + bias[jj]);
      float fg = sigm(gr[Hz + jj] + bias[Hz + jj]);
      float gg = tanhf(gr[2 * Hz + jj] + bias[2 * Hz + jj]);
      float og = sigm(gr[3 * Hz + jj] + bias[3 * Hz + jj]);
      float c = fg * cst[idx] + ig * gg;
      cst[idx] = c;
      float h = og * tanhf(c);
      hst[idx] = f2bf(h);
      if (yout) yout[(size_t)t * Bz * Hz + idx] = f2bf(h);
      if (hfinal && t == Sz - 1) hfinal[idx] = h;
    }
    __syncthreads();
  }
}

__global__ void __launch_bounds__(256, 1)
lstm_kernel(const __bf16* __restrict__ xseq,
            const __bf16* __restrict__ w0ih, const __bf16* __restrict__ w0hh,
            const float* __restrict__ bih0, const float* __restrict__ bhh0,
            const __bf16* __restrict__ w1ih, const __bf16* __restrict__ w1hh,
            const float* __restrict__ bih1, const float* __restrict__ bhh1,
            __bf16* __restrict__ y0, float* __restrict__ price_repr) {
  __shared__ float  gates[Bz * G4H];   // 32 KB
  __shared__ float  cst[Bz * Hz];      // 8 KB
  __shared__ __bf16 hst[Bz * Hz];      // 4 KB
  __shared__ float  bias[G4H];         // 2 KB
  int tid = threadIdx.x;
  run_layer<2>(xseq, w0ih, w0hh, bih0, bhh0, y0, nullptr, gates, cst, hst, bias, tid);
  __threadfence_block();
  __syncthreads();
  run_layer<4>(y0, w1ih, w1hh, bih1, bhh1, nullptr, price_repr, gates, cst, hst, bias, tid);
}

// ================= attention: rank-1 scores + softmax + attn write ===========
__device__ inline float wred_max(float v) {
#pragma unroll
  for (int o = 16; o > 0; o >>= 1) v = fmaxf(v, __shfl_xor(v, o, 32));
  return v;
}
__device__ inline float wred_sum(float v) {
#pragma unroll
  for (int o = 16; o > 0; o >>= 1) v += __shfl_xor(v, o, 32);
  return v;
}

__global__ void __launch_bounds__(256)
attn_kernel(const float* __restrict__ g, const float* __restrict__ mask,
            const float* __restrict__ adjw, const float* __restrict__ ab,
            float* __restrict__ out, float* __restrict__ s_out) {
  __shared__ float red[3][8];
  int b = blockIdx.x >> 10, i = blockIdx.x & (Nz - 1);
  int tid = threadIdx.x;
  const float* gb = g + b * Nz;
  float base = ab[0] * gb[i];
  float beta = ab[1];
  const float* mrow = mask + (size_t)i * Nz;
  const float* wrow = adjw + (size_t)i * Nz;
  float sc[4], gj[4];
  float mx = -3.4e38f;
#pragma unroll
  for (int r = 0; r < 4; r++) {
    int j = tid + r * 256;
    float gv = gb[j]; gj[r] = gv;
    float w = mrow[j] * sigm(wrow[j]);
    float s = (base + beta * gv) * w;
    sc[r] = s;
    mx = fmaxf(mx, s);
  }
  mx = wred_max(mx);
  if ((tid & 31) == 0) red[0][tid >> 5] = mx;
  __syncthreads();
  float bm = red[0][0];
#pragma unroll
  for (int k = 1; k < 8; k++) bm = fmaxf(bm, red[0][k]);
  float sum = 0.f, dot = 0.f;
#pragma unroll
  for (int r = 0; r < 4; r++) {
    float e = __expf(sc[r] - bm);
    sum += e;
    dot += e * gj[r];
    sc[r] = e;
  }
  sum = wred_sum(sum);
  dot = wred_sum(dot);
  if ((tid & 31) == 0) { red[1][tid >> 5] = sum; red[2][tid >> 5] = dot; }
  __syncthreads();
  float ts = 0.f, td = 0.f;
#pragma unroll
  for (int k = 0; k < 8; k++) { ts += red[1][k]; td += red[2][k]; }
  float inv = 1.f / ts;
  float* orow = out + 16 + ((size_t)(b * Nz + i)) * Nz;
#pragma unroll
  for (int r = 0; r < 4; r++) orow[tid + r * 256] = sc[r] * inv;
  if (tid == 0) s_out[b * Nz + i] = td * inv;
}

// ================= pack fused-GEMM operands (bf16) ===========================
__global__ void pack_kernel(const float* __restrict__ pr, const float* __restrict__ sb,
                            const float* __restrict__ Wf, const float* __restrict__ WfG,
                            __bf16* __restrict__ CA, __bf16* __restrict__ Bt) {
  int tid = blockIdx.x * blockDim.x + threadIdx.x;
  int stride = gridDim.x * blockDim.x;
  for (int idx = tid; idx < Bz * KT; idx += stride) {          // A = [price_repr | s]
    int b = idx / KT, k = idx % KT;
    float v = (k < Hz) ? pr[b * Hz + k] : sb[b * Nz + (k - Hz)];
    CA[idx] = f2bf(v);
  }
  for (int idx = tid; idx < FUSEz * KT; idx += stride) {       // Bt[f][k] = [Wf^T | WfG^T]
    int f = idx / KT, k = idx % KT;
    float v = (k < Hz) ? Wf[(size_t)k * FUSEz + f] : WfG[(size_t)(k - Hz) * FUSEz + f];
    Bt[idx] = f2bf(v);
  }
}

// ================= fused head: GEMM(16x64,K=1152) + MLP + predictions ========
__global__ void __launch_bounds__(128)
fuse_kernel(const __bf16* __restrict__ CA, const __bf16* __restrict__ Bt,
            const float* __restrict__ bfv, const float* __restrict__ W1,
            const float* __restrict__ b1, const float* __restrict__ W2,
            const float* __restrict__ b2, float* __restrict__ out) {
  __shared__ float fused[Bz][FUSEz];
  __shared__ float hid[Bz][H1z];
  int tid = threadIdx.x;
  int wave = tid >> 5, lane = tid & 31;
  v8f acc = {};
  for (int kc = 0; kc < KT / 32; kc++) {
    v16bf a  = load_atile(CA + kc * 32, KT, lane);
    v16bf bb = load_btile(Bt, KT, wave * 16, kc * 32, lane);
    acc = wmma_bf16(a, bb, acc);
  }
  int n  = wave * 16 + (lane & 15);
  int mb = (lane >> 4) * 8;
#pragma unroll
  for (int r = 0; r < 8; r++) fused[mb + r][n] = fmaxf(0.f, acc[r] + bfv[n]);
  __syncthreads();
  for (int idx = tid; idx < Bz * H1z; idx += 128) {
    int b = idx >> 5, hh = idx & 31;
    float a2 = b1[hh];
    for (int f = 0; f < FUSEz; f++) a2 += fused[b][f] * W1[f * H1z + hh];
    hid[b][hh] = fmaxf(0.f, a2);
  }
  __syncthreads();
  if (tid < Bz) {
    float p = b2[0];
    for (int hh = 0; hh < H1z; hh++) p += hid[tid][hh] * W2[hh];
    out[tid] = p;
  }
}

// ================= host launcher =============================================
extern "C" void kernel_launch(void* const* d_in, const int* in_sizes, int n_in,
                              void* d_out, int out_size, void* d_ws, size_t ws_size,
                              hipStream_t stream) {
  const float* price = (const float*)d_in[0];
  const float* gfeat = (const float*)d_in[1];
  const float* amask = (const float*)d_in[2];
  const float* Wih0  = (const float*)d_in[3];
  const float* Whh0  = (const float*)d_in[4];
  const float* bih0  = (const float*)d_in[5];
  const float* bhh0  = (const float*)d_in[6];
  const float* Wih1  = (const float*)d_in[7];
  const float* Whh1  = (const float*)d_in[8];
  const float* bih1  = (const float*)d_in[9];
  const float* bhh1  = (const float*)d_in[10];
  const float* Wg    = (const float*)d_in[11];
  const float* a_src = (const float*)d_in[12];
  const float* a_dst = (const float*)d_in[13];
  const float* adjw  = (const float*)d_in[14];
  const float* Wf    = (const float*)d_in[15];
  const float* bfv   = (const float*)d_in[16];
  const float* W1    = (const float*)d_in[17];
  const float* b1    = (const float*)d_in[18];
  const float* W2    = (const float*)d_in[19];
  const float* b2    = (const float*)d_in[20];
  (void)in_sizes; (void)n_in; (void)out_size; (void)ws_size;

  char* ws = (char*)d_ws;
  __bf16* xseq = (__bf16*)(ws + OFF_XSEQ);
  __bf16* y0   = (__bf16*)(ws + OFF_Y0);
  __bf16* w0ih = (__bf16*)(ws + OFF_W0IH);
  __bf16* w0hh = (__bf16*)(ws + OFF_W0HH);
  __bf16* w1ih = (__bf16*)(ws + OFF_W1IH);
  __bf16* w1hh = (__bf16*)(ws + OFF_W1HH);
  float*  pr   = (float*)(ws + OFF_PR);
  float*  sbuf = (float*)(ws + OFF_S);
  float*  WfG  = (float*)(ws + OFF_WFG);
  float*  ab   = (float*)(ws + OFF_AB);
  __bf16* CA   = (__bf16*)(ws + OFF_CA);
  __bf16* Bt   = (__bf16*)(ws + OFF_BT);
  float*  out  = (float*)d_out;

  prep_kernel<<<512, 256, 0, stream>>>(price, Wih0, Whh0, Wih1, Whh1, Wg, a_src, a_dst,
                                       xseq, w0ih, w0hh, w1ih, w1hh, ab);
  wfg_kernel<<<Nz, FUSEz, 0, stream>>>(Wg, Wf, WfG);
  lstm_kernel<<<1, 256, 0, stream>>>(xseq, w0ih, w0hh, bih0, bhh0,
                                     w1ih, w1hh, bih1, bhh1, y0, pr);
  attn_kernel<<<Bz * Nz, 256, 0, stream>>>(gfeat, amask, adjw, ab, out, sbuf);
  pack_kernel<<<360, 256, 0, stream>>>(pr, sbuf, Wf, WfG, CA, Bt);
  fuse_kernel<<<1, 128, 0, stream>>>(CA, Bt, bfv, W1, b1, W2, b2, out);
}
